// VTS_5849745457575
// MI455X (gfx1250) — compile-verified, matrix operand order checked by reading, not run on hardware
//
#include <hip/hip_runtime.h>
#include <cstdint>

#define STR2(x) #x
#define STR1(x) STR2(x)
#pragma message("gfx1250 build: clang " STR1(__clang_major__) "." STR1(__clang_minor__))

// ---------------- problem constants ----------------
#define Bb   16
#define Ll   512
#define Dd   2048
#define Hh   512
#define OMh  5
#define Tt   502            // L - 2*OM
#define Mrows (Tt*Bb)       // 8032
#define G4   2048           // 4*H
#define KSEQ 544            // 513 padded to 17*32

// ---------------- WMMA fragment types ----------------
typedef __attribute__((ext_vector_type(16))) __bf16 v16bf;
typedef __attribute__((ext_vector_type(8)))  float  v8f;
typedef __attribute__((ext_vector_type(4)))  unsigned int u32x4;
typedef __attribute__((ext_vector_type(8)))  int  i32x8;
typedef __attribute__((ext_vector_type(4)))  int  i32x4;

union Frag {
    uint32_t u[8];
    uint4    q[2];
    v16bf    v;
};

__device__ __forceinline__ uint32_t f2bf1(float f) {
    uint32_t u = __float_as_uint(f);
    return (u + 0x7FFFu + ((u >> 16) & 1u)) >> 16;   // RNE
}
__device__ __forceinline__ uint32_t pack2bf(float a, float b) {
    return f2bf1(a) | (f2bf1(b) << 16);
}
__device__ __forceinline__ float sigm(float x) { return 1.f / (1.f + expf(-x)); }

// ---------------- elementwise f32 -> bf16 (4 per thread) ----------------
__global__ void __launch_bounds__(256) k_f32_to_bf16(const float* __restrict__ in,
                                                     uint32_t* __restrict__ out, int n4) {
    int i = blockIdx.x * blockDim.x + threadIdx.x;
    if (i < n4) {
        float4 p = ((const float4*)in)[i];
        uint2 o;
        o.x = pack2bf(p.x, p.y);
        o.y = pack2bf(p.z, p.w);
        ((uint2*)out)[i] = o;
    }
}

// ---------------- pack weights into WMMA B-fragment layout ----------------
// layout: [ntile][ktile][lane][v] dwords; dword = bf16 pair (k0, k0+1) for column n
// mode 0: conv weight src[h,d,koff] (N, 2048, KK); W2[k][n] = src[(n*2048 + (k&2047))*KK + (k>>11)]
// mode 1: dense row-major src (N, Ksrc);           W2[k][n] = k<Ksrc ? src[n*Ksrc+k] : 0
__global__ void __launch_bounds__(256) k_pack_b(const float* __restrict__ src,
                                                uint32_t* __restrict__ out,
                                                int Ktiles, int Ntiles, int mode, int KK, int Ksrc) {
    int gid = blockIdx.x * blockDim.x + threadIdx.x;
    int total = Ktiles * Ntiles * 256;
    if (gid >= total) return;
    int v    = gid & 7;
    int lane = (gid >> 3) & 31;
    int tile = gid >> 8;
    int kt = tile % Ktiles;
    int nt = tile / Ktiles;
    int n    = nt * 16 + (lane & 15);
    int half = lane >> 4;
    int k0 = kt * 32 + ((v < 4) ? 2 * v : 16 + 2 * (v - 4)) + 8 * half;
    float e0, e1;
    if (mode == 0) {
        e0 = src[(n * Dd + (k0 & (Dd - 1))) * KK + (k0 >> 11)];
        int k1 = k0 + 1;
        e1 = src[(n * Dd + (k1 & (Dd - 1))) * KK + (k1 >> 11)];
    } else {
        e0 = (k0     < Ksrc) ? src[n * Ksrc + k0]     : 0.f;
        e1 = (k0 + 1 < Ksrc) ? src[n * Ksrc + k0 + 1] : 0.f;
    }
    out[gid] = pack2bf(e0, e1);
}

// ---------------- register-blocked WMMA GEMM: 2 M-tiles x 4 N-tiles per wave ----------------
// A row m = t*16+b reads elements starting at  b*strideB + t*strideT + offA
// 12 b128 loads per 8 wmma (A frags reused 4x, B frags 2x)
__global__ void __launch_bounds__(256) k_gemm(const uint16_t* __restrict__ A,
                                              const uint32_t* __restrict__ Bp,
                                              const float* __restrict__ bias0,
                                              const float* __restrict__ bias1,
                                              float* __restrict__ C,
                                              int Mtiles, int Ntiles, int Ktiles,
                                              long strideB, long strideT, long offA, int ldC) {
    int wave = (blockIdx.x * blockDim.x + threadIdx.x) >> 5;
    int lane = threadIdx.x & 31;
    int Mq = Mtiles >> 1, Nq = Ntiles >> 2;
    if (wave >= Mq * Nq) return;
    int tn = (wave % Nq) * 4;
    int tm = (wave / Nq) * 2;
    int half = lane >> 4;
    int r    = lane & 15;
    const uint32_t* a32 = (const uint32_t*)A;
    size_t arow0 = ((size_t)r * strideB + (size_t)tm * strideT + (size_t)offA) >> 1;       // dwords
    size_t arow1 = ((size_t)r * strideB + (size_t)(tm + 1) * strideT + (size_t)offA) >> 1;
    size_t nstride = (size_t)Ktiles * 256;
    const uint32_t* bp = Bp + ((size_t)tn * Ktiles) * 256 + lane * 8;
    v8f acc[2][4];
#pragma unroll
    for (int mi = 0; mi < 2; ++mi)
#pragma unroll
        for (int j = 0; j < 4; ++j) acc[mi][j] = (v8f){0, 0, 0, 0, 0, 0, 0, 0};
    for (int kt = 0; kt < Ktiles; ++kt) {
        Frag a0, a1, b[4];
        size_t ai0 = arow0 + (size_t)kt * 16 + 4 * half;
        size_t ai1 = arow1 + (size_t)kt * 16 + 4 * half;
        a0.q[0] = *(const uint4*)(a32 + ai0);
        a0.q[1] = *(const uint4*)(a32 + ai0 + 8);
        a1.q[0] = *(const uint4*)(a32 + ai1);
        a1.q[1] = *(const uint4*)(a32 + ai1 + 8);
        const uint32_t* bq = bp;
#pragma unroll
        for (int j = 0; j < 4; ++j) {
            b[j].q[0] = *(const uint4*)(bq);
            b[j].q[1] = *(const uint4*)(bq + 4);
            bq += nstride;
        }
        __builtin_prefetch((const void*)(bp + 512), 0, 1);   // global_prefetch_b8 (B stream)
        bp += 256;
#pragma unroll
        for (int j = 0; j < 4; ++j) {
            acc[0][j] = __builtin_amdgcn_wmma_f32_16x16x32_bf16(false, a0.v, false, b[j].v,
                                                                (short)0, acc[0][j], false, false);
            acc[1][j] = __builtin_amdgcn_wmma_f32_16x16x32_bf16(false, a1.v, false, b[j].v,
                                                                (short)0, acc[1][j], false, false);
        }
    }
#pragma unroll
    for (int mi = 0; mi < 2; ++mi)
#pragma unroll
        for (int j = 0; j < 4; ++j) {
            int n = (tn + j) * 16 + r;
            float bsum = (bias0 ? bias0[n] : 0.f) + (bias1 ? bias1[n] : 0.f);
#pragma unroll
            for (int v = 0; v < 8; ++v) {
                int m = (tm + mi) * 16 + v + 8 * half;   // C layout: VGPR v -> row v + 8*(lane>=16)
                C[(size_t)m * ldC + n] = acc[mi][j][v] + bsum;
            }
        }
}

// ---------------- vts_d reduction + build padded bf16 seq (8032 x 544) ----------------
__global__ void __launch_bounds__(256) k_build_seq(const float* __restrict__ db,
                                                   const float* __restrict__ da,
                                                   const float* __restrict__ rr,
                                                   uint32_t* __restrict__ seq) {
    int wave = (blockIdx.x * blockDim.x + threadIdx.x) >> 5;
    int lane = threadIdx.x & 31;
    if (wave >= Mrows) return;
    size_t row = (size_t)wave * Hh;
    float s = 0.f;
#pragma unroll
    for (int i = 0; i < 16; ++i) {
        int c = lane * 16 + i;
        s += db[row + c] * da[row + c];
    }
    for (int off = 16; off; off >>= 1) s += __shfl_xor(s, off, 32);
    size_t srow = (size_t)wave * (KSEQ / 2);                 // dword row
#pragma unroll
    for (int i = 0; i < 8; ++i) {
        int c = lane * 16 + 2 * i;
        seq[srow + lane * 8 + i] = pack2bf(rr[row + c], rr[row + c + 1]);
    }
    if (lane < 16)                                           // cols 512..543: vts_d then zero pad
        seq[srow + 256 + lane] = (lane == 0) ? f2bf1(s) : 0u;
}

// ---------------- zero h double-buffers + barrier counters ----------------
__global__ void __launch_bounds__(256) k_init(uint32_t* __restrict__ hbuf, int n, int* __restrict__ cnt) {
    int i = blockIdx.x * blockDim.x + threadIdx.x;
    if (i < n) hbuf[i] = 0;
    if (i < 2) cnt[i] = 0;
}

// ---------------- persistent bi-LSTM recurrence ----------------
// 32 WGs: dir = blockIdx.x>>4, wg = blockIdx.x&15; 64 threads (2 waves).
// Each WG owns 32 h-columns; its whh slice (512 x 128 bf16 = 128KB) is TDM-loaded into LDS
// (CDNA5: 320KB/WGP). h exchanged via double-buffered global bf16 slab + device-scope atomic
// barrier; c stays in registers.
#define NWG 16
__global__ void __launch_bounds__(64) k_lstm(const float* __restrict__ gx_f,
                                             const float* __restrict__ gx_r,
                                             const uint32_t* __restrict__ whhP_f,
                                             const uint32_t* __restrict__ whhP_r,
                                             uint32_t* __restrict__ hbuf,   // [dir][slot][4096] dwords
                                             float* __restrict__ hs_f, float* __restrict__ hs_r,
                                             int* __restrict__ cnt) {
    extern __shared__ uint32_t ldsw[];                       // 8 tiles * 4096 dwords = 128KB
    int dir = blockIdx.x >> 4;
    int wg  = blockIdx.x & 15;
    int tid = threadIdx.x;
    int wv  = tid >> 5;
    int lane = tid & 31;
    int half = lane >> 4;
    const float*    gx     = dir ? gx_r   : gx_f;
    const uint32_t* wp_src = dir ? whhP_r : whhP_f;
    float*          hs     = dir ? hs_r   : hs_f;

    // ---- stage whh slice into LDS: 4 contiguous 32KB chunks (gate g -> ntiles g*32+wg*2, +1) ----
#if __has_builtin(__builtin_amdgcn_tensor_load_to_lds)
    if (wv == 0) {                                           // TDM is per-wave; issue once
        uint32_t lds_base = (uint32_t)(uintptr_t)(void*)ldsw;
        for (int g = 0; g < 4; ++g) {
            const uint32_t* gsrc = wp_src + (size_t)(g * 32 + wg * 2) * 4096;
            uint64_t ga = (uint64_t)(uintptr_t)gsrc;
            // D# group0: count=1, lds_addr, global_addr[56:0], type=2 (ISA 8.3)
            u32x4 g0 = { 1u,
                         lds_base + (uint32_t)g * 32768u,
                         (uint32_t)ga,
                         (uint32_t)((ga >> 32) & 0x01FFFFFFu) | (2u << 30) };
            // D# group1: mask=0, data_size=4B(2), tensor_dim0=8192, tensor_dim1=1,
            //            tile_dim0=8192, tile_dim1=1, dim0_stride=8192 (ISA 8.4)
            i32x8 g1 = { (int)(2u << 16),
                         (int)(8192u << 16),
                         (int)(1u << 16),
                         (int)(8192u << 16),
                         1, 8192, 0, 0 };
            i32x4 z4 = {0, 0, 0, 0};
#if defined(__clang_major__) && (__clang_major__ >= 23)
            i32x8 z8 = {0, 0, 0, 0, 0, 0, 0, 0};
            __builtin_amdgcn_tensor_load_to_lds(g0, g1, z4, z4, z8, 0);
#else
            __builtin_amdgcn_tensor_load_to_lds(g0, g1, z4, z4, 0);
#endif
        }
        __builtin_amdgcn_s_wait_tensorcnt(0);
    }
#else
    for (int i = tid; i < 8 * 4096; i += 64) {
        int tileIdx = i >> 12, rest = i & 4095;
        int g = tileIdx >> 1, sub = tileIdx & 1;
        ldsw[i] = wp_src[(size_t)((g * 32 + wg * 2 + sub) * 4096 + rest)];
    }
#endif
    __syncthreads();

    uint32_t* hb0 = hbuf + (size_t)dir * 8192;
    int* mycnt = cnt + dir;
    float c_reg[8];
#pragma unroll
    for (int v = 0; v < 8; ++v) c_reg[v] = 0.f;
    int hcol = wg * 32 + wv * 16 + (lane & 15);
    for (int s = 0; s < Tt; ++s) {
        int t = dir ? (Tt - 1 - s) : s;
        const uint32_t* hcur = hb0 + (size_t)(s & 1) * 4096;
        uint32_t*       hnxt = hb0 + (size_t)((s + 1) & 1) * 4096;
        v8f acc[4];
#pragma unroll
        for (int g = 0; g < 4; ++g) acc[g] = (v8f){0, 0, 0, 0, 0, 0, 0, 0};
        for (int kt = 0; kt < 16; ++kt) {                    // A(h) frag loaded once per kt
            Frag a;
            size_t ai = (size_t)(lane & 15) * 256 + kt * 16 + 4 * half;
            a.q[0] = *(const uint4*)(hcur + ai);
            a.q[1] = *(const uint4*)(hcur + ai + 8);
#pragma unroll
            for (int g = 0; g < 4; ++g) {                    // B from LDS: ds_read_b128
                Frag b;
                const uint32_t* wp = ldsw + ((g * 2 + wv) << 12) + kt * 256 + lane * 8;
                b.q[0] = *(const uint4*)(wp);
                b.q[1] = *(const uint4*)(wp + 4);
                acc[g] = __builtin_amdgcn_wmma_f32_16x16x32_bf16(false, a.v, false, b.v,
                                                                 (short)0, acc[g], false, false);
            }
        }
        const float* gxr = gx + (size_t)t * Bb * G4;
#pragma unroll
        for (int v = 0; v < 8; ++v) {
            int bi = v + 8 * half;                           // batch index from C layout
            const float* gb = gxr + (size_t)bi * G4 + hcol;
            float gi = gb[0]    + acc[0][v];
            float gf = gb[512]  + acc[1][v];
            float gg = gb[1024] + acc[2][v];
            float go = gb[1536] + acc[3][v];
            float cc = sigm(gf) * c_reg[v] + sigm(gi) * tanhf(gg);
            c_reg[v] = cc;
            float hh = sigm(go) * tanhf(cc);
            hs[((size_t)t * Bb + bi) * Hh + hcol] = hh;
            ((uint16_t*)hnxt)[bi * Hh + hcol] = (uint16_t)f2bf1(hh);
        }
        // device-scope release/acquire barrier across the 16 WGs of this direction
        __threadfence();
        __syncthreads();
        if (tid == 0) {
            atomicAdd(mycnt, 1);
            while (__hip_atomic_load(mycnt, __ATOMIC_ACQUIRE, __HIP_MEMORY_SCOPE_AGENT) < NWG * (s + 1))
                __builtin_amdgcn_s_sleep(1);
        }
        __syncthreads();
        __threadfence();
    }
}

// ---------------- final FC + sigmoid, out (B,T) ----------------
__global__ void __launch_bounds__(256) k_fc(const float* __restrict__ hsf,
                                            const float* __restrict__ hsr,
                                            const float* __restrict__ wfc,
                                            const float* __restrict__ bfc,
                                            float* __restrict__ out) {
    int wave = (blockIdx.x * blockDim.x + threadIdx.x) >> 5;
    int lane = threadIdx.x & 31;
    if (wave >= Mrows) return;
    size_t row = (size_t)wave * Hh;
    float acc = 0.f;
    for (int h = lane; h < Hh; h += 32)
        acc += hsf[row + h] * wfc[h] + hsr[row + h] * wfc[Hh + h];
    for (int off = 16; off; off >>= 1) acc += __shfl_xor(acc, off, 32);
    if (lane == 0) {
        int t = wave >> 4, b = wave & 15;
        out[(size_t)b * Tt + t] = sigm(acc + bfc[0]);
    }
}

// ---------------- host orchestration ----------------
extern "C" void kernel_launch(void* const* d_in, const int* in_sizes, int n_in,
                              void* d_out, int out_size, void* d_ws, size_t ws_size,
                              hipStream_t stream) {
    (void)in_sizes; (void)n_in; (void)out_size; (void)ws_size;
    const float* x     = (const float*)d_in[0];
    const float* wb    = (const float*)d_in[1];
    const float* bb    = (const float*)d_in[2];
    const float* wa    = (const float*)d_in[3];
    const float* ba    = (const float*)d_in[4];
    const float* wr    = (const float*)d_in[5];
    const float* br    = (const float*)d_in[6];
    const float* wih_f = (const float*)d_in[7];
    const float* whh_f = (const float*)d_in[8];
    const float* bih_f = (const float*)d_in[9];
    const float* bhh_f = (const float*)d_in[10];
    const float* wih_r = (const float*)d_in[11];
    const float* whh_r = (const float*)d_in[12];
    const float* bih_r = (const float*)d_in[13];
    const float* bhh_r = (const float*)d_in[14];
    const float* wfc   = (const float*)d_in[15];
    const float* bfc   = (const float*)d_in[16];
    float* out = (float*)d_out;

    char* ws = (char*)d_ws;
    size_t off = 0;
    auto alloc = [&](size_t bytes) -> char* {
        char* p = ws + off;
        off += (bytes + 255) & ~(size_t)255;
        return p;
    };
    uint32_t* x_bf = (uint32_t*)alloc((size_t)Bb * Ll * Dd * 2);
    uint32_t* wbP  = (uint32_t*)alloc((size_t)320 * 32 * 256 * 4);
    uint32_t* waP  = (uint32_t*)alloc((size_t)320 * 32 * 256 * 4);
    uint32_t* wrP  = (uint32_t*)alloc((size_t)640 * 32 * 256 * 4);
    uint32_t* wifP = (uint32_t*)alloc((size_t)17 * 128 * 256 * 4);
    uint32_t* wirP = (uint32_t*)alloc((size_t)17 * 128 * 256 * 4);
    uint32_t* whfP = (uint32_t*)alloc((size_t)16 * 128 * 256 * 4);
    uint32_t* whrP = (uint32_t*)alloc((size_t)16 * 128 * 256 * 4);
    float* dbuf = (float*)alloc((size_t)Mrows * Hh * 4);
    float* abuf = (float*)alloc((size_t)Mrows * Hh * 4);
    float* rbuf = (float*)alloc((size_t)Mrows * Hh * 4);
    uint32_t* seq = (uint32_t*)alloc((size_t)Mrows * (KSEQ / 2) * 4);
    float* gxf = (float*)alloc((size_t)Mrows * G4 * 4);
    float* gxr = (float*)alloc((size_t)Mrows * G4 * 4);
    float* hsf = (float*)alloc((size_t)Mrows * Hh * 4);
    float* hsr = (float*)alloc((size_t)Mrows * Hh * 4);
    uint32_t* hbuf = (uint32_t*)alloc((size_t)16384 * 4);
    int* cnt = (int*)alloc(256);

    // 1) x -> bf16
    {
        int n4 = Bb * Ll * Dd / 4;
        k_f32_to_bf16<<<(n4 + 255) / 256, 256, 0, stream>>>(x, x_bf, n4);
    }
    // 2) pack weights into WMMA B-fragment layout
    auto packConv = [&](const float* src, uint32_t* dst, int KK) {
        int Kt = KK * Dd / 32, total = Kt * 32 * 256;
        k_pack_b<<<(total + 255) / 256, 256, 0, stream>>>(src, dst, Kt, 32, 0, KK, 0);
    };
    packConv(wb, wbP, 5);
    packConv(wa, waP, 5);
    packConv(wr, wrP, 10);
    auto packDense = [&](const float* src, uint32_t* dst, int Kt, int Nt, int Ksrc) {
        int total = Kt * Nt * 256;
        k_pack_b<<<(total + 255) / 256, 256, 0, stream>>>(src, dst, Kt, Nt, 1, 0, Ksrc);
    };
    packDense(wih_f, wifP, 17, 128, 513);
    packDense(wih_r, wirP, 17, 128, 513);
    packDense(whh_f, whfP, 16, 128, 512);
    packDense(whh_r, whrP, 16, 128, 512);
    // 3) conv GEMMs (A rows overlap inside x: row (b,t) = x[b, t+off .. t+off+KK, :] contiguous)
    {
        long sB = (long)Ll * Dd, sT = Dd;
        int waves = (502 / 2) * (32 / 4);                       // 2x4 tiles per wave
        int blocks = (waves * 32 + 255) / 256;
        k_gemm<<<blocks, 256, 0, stream>>>((const uint16_t*)x_bf, wbP, bb, nullptr, dbuf,
                                           502, 32, 320, sB, sT, 0, Hh);
        k_gemm<<<blocks, 256, 0, stream>>>((const uint16_t*)x_bf, waP, ba, nullptr, abuf,
                                           502, 32, 320, sB, sT, (long)OMh * Dd, Hh);
        k_gemm<<<blocks, 256, 0, stream>>>((const uint16_t*)x_bf, wrP, br, nullptr, rbuf,
                                           502, 32, 640, sB, sT, 0, Hh);
    }
    // 4) vts_d + seq (t,b,513 padded 544) in bf16
    k_build_seq<<<Mrows / 8, 256, 0, stream>>>(dbuf, abuf, rbuf, seq);
    // 5) input-gate precompute for both directions (hoisted out of the recurrence)
    {
        long sB = KSEQ, sT = (long)KSEQ * Bb;
        int waves = (502 / 2) * (128 / 4);
        int blocks = (waves * 32 + 255) / 256;
        k_gemm<<<blocks, 256, 0, stream>>>((const uint16_t*)seq, wifP, bih_f, bhh_f, gxf,
                                           502, 128, 17, sB, sT, 0, G4);
        k_gemm<<<blocks, 256, 0, stream>>>((const uint16_t*)seq, wirP, bih_r, bhh_r, gxr,
                                           502, 128, 17, sB, sT, 0, G4);
    }
    // 6) recurrence: 2 dirs x 16 WGs, 128KB dynamic LDS each (TDM-staged whh slice)
    k_init<<<(16384 + 255) / 256, 256, 0, stream>>>(hbuf, 16384, cnt);
    k_lstm<<<32, 64, 8 * 4096 * 4, stream>>>(gxf, gxr, whfP, whrP, hbuf, hsf, hsr, cnt);
    // 7) final FC + sigmoid
    k_fc<<<Mrows * 32 / 256, 256, 0, stream>>>(hsf, hsr, wfc, bfc, out);
}
